// ConceptEmbedding_83657372991873
// MI455X (gfx1250) — matrix operand order repaired and checked by reading, not compiled, not run on hardware
//
#include <hip/hip_runtime.h>
#include <cstdint>

#define B_DIM   2048
#define IN_DIM  1024
#define K_DIM   128
#define E_DIM   64
#define N_DIM   128   // 2*E

typedef __attribute__((ext_vector_type(16))) _Float16 v16h;
typedef __attribute__((ext_vector_type(8)))  _Float16 h8;
typedef __attribute__((ext_vector_type(4)))  _Float16 h4;
typedef __attribute__((ext_vector_type(2)))  __fp16   pk2;   // cvt_pkrtz result type
typedef __attribute__((ext_vector_type(8)))  float    v8f;

union H16 { v16h v; h8 h[2]; };
union H4  { h4 v;  pk2 p[2]; };

#define LDA 40    // LDS row stride in halfs for As/Bt: 80B rows, 16B aligned groups
#define CLD 130   // LDS row stride in floats for ctx

__launch_bounds__(256)
__global__ void concept_embed_kernel(const float* __restrict__ x,
                                     const float* __restrict__ W,
                                     const float* __restrict__ bvec,
                                     const float* __restrict__ wp,
                                     const float* __restrict__ bp,
                                     float* __restrict__ cemb,
                                     float* __restrict__ cpred) {
  __shared__ __align__(16) _Float16 As[128 * LDA];   // x tile, row-major (M x Kc)
  __shared__ __align__(16) _Float16 Bt[128 * LDA];   // W tile, N-major  (N x Kc)
  __shared__ float ctx[128 * CLD];                   // post-activation context tile
  __shared__ float pbuf[128];
  __shared__ float bs[128];
  __shared__ float wps[128];

  const int tid  = threadIdx.x;
  const int lane = tid & 31;
  const int wave = tid >> 5;        // 0..7 -> 16-row M stripe
  const int m0   = blockIdx.x * 128;
  const int kb   = blockIdx.y;      // concept index k
  const float* __restrict__ Wk = W + (size_t)kb * IN_DIM * N_DIM;

  if (tid < 128) {
    bs[tid]  = bvec[kb * N_DIM + tid];
    wps[tid] = wp[tid];
  }

  v8f acc[8];
#pragma unroll
  for (int t = 0; t < 8; ++t) acc[t] = (v8f)0.0f;

  const int mrow = wave * 16 + (lane & 15);   // A-fragment row
  const int ka   = (lane >> 4) << 3;          // A-fragment K offset: 0 or 8
  const int kofB = (lane >> 4) << 3;          // B-fragment K offset: 0 or 8

  for (int c0 = 0; c0 < IN_DIM; c0 += 32) {
    // ---- stage x chunk (128 x 32 f32 -> f16), b128 reads + packed cvt ----
#pragma unroll
    for (int it = 0; it < 4; ++it) {
      const int idx4 = tid + it * 256;           // 0..1023 float4 slots
      const int r    = idx4 >> 3;                // row 0..127
      const int c4   = (idx4 & 7) << 2;          // col 0..28 step 4
      const float4 v = *(const float4*)(x + (size_t)(m0 + r) * IN_DIM + c0 + c4);
      H4 hv;
      hv.p[0] = __builtin_amdgcn_cvt_pkrtz(v.x, v.y);
      hv.p[1] = __builtin_amdgcn_cvt_pkrtz(v.z, v.w);
      *(h4*)(As + r * LDA + c4) = hv.v;
    }
    // ---- stage W chunk transposed: Bt[n][kk] = W[k][c0+kk][n] ----
#pragma unroll
    for (int it = 0; it < 4; ++it) {
      const int idx4 = tid + it * 256;           // 0..1023 float4 slots
      const int kk   = idx4 >> 5;                // 0..31
      const int n4   = (idx4 & 31) << 2;         // 0..124 step 4
      const float4 v = *(const float4*)(Wk + (size_t)(c0 + kk) * N_DIM + n4);
      Bt[(n4 + 0) * LDA + kk] = (_Float16)v.x;
      Bt[(n4 + 1) * LDA + kk] = (_Float16)v.y;
      Bt[(n4 + 2) * LDA + kk] = (_Float16)v.z;
      Bt[(n4 + 3) * LDA + kk] = (_Float16)v.w;
    }
    // prefetch next K-chunk into cache while we compute this one
    if (c0 + 32 < IN_DIM) {
      __builtin_prefetch(x  + (size_t)(m0 + (tid >> 1)) * IN_DIM + c0 + 32 + ((tid & 1) << 4), 0, 1);
      __builtin_prefetch(Wk + (size_t)(c0 + 32 + (tid >> 3)) * N_DIM + ((tid & 7) << 4), 0, 1);
    }
    __syncthreads();

    // ---- batch ALL fragment loads first so DS latency overlaps the WMMA chain ----
    H16 a;
    a.h[0] = *(const h8*)(As + mrow * LDA + ka);
    a.h[1] = *(const h8*)(As + mrow * LDA + ka + 16);

    H16 bf[8];
#pragma unroll
    for (int t = 0; t < 8; ++t) {
      const int ncol = t * 16 + (lane & 15);
      bf[t].h[0] = *(const h8*)(Bt + ncol * LDA + kofB);
      bf[t].h[1] = *(const h8*)(Bt + ncol * LDA + kofB + 16);
    }

#pragma unroll
    for (int t = 0; t < 8; ++t) {
      acc[t] = __builtin_amdgcn_wmma_f32_16x16x32_f16(
          /*neg_a=*/false, a.v, /*neg_b=*/false, bf[t].v,
          /*c_mod=*/(short)0, acc[t], /*reuse_a=*/false, /*reuse_b=*/false);
    }
    __syncthreads();
  }

  // ---- epilogue: bias + leaky_relu(0.01) -> ctx in LDS ----
  const int mloc = wave * 16 + ((lane >> 4) << 3);
#pragma unroll
  for (int t = 0; t < 8; ++t) {
    const int n = t * 16 + (lane & 15);
    const float bn = bs[n];
#pragma unroll
    for (int r = 0; r < 8; ++r) {
      float v = acc[t][r] + bn;
      v = (v > 0.0f) ? v : 0.01f * v;
      ctx[(mloc + r) * CLD + n] = v;
    }
  }
  __syncthreads();

  // ---- gate: c_pred[m] = sigmoid(ctx[m,:] . wp + bp) ----
  const float bpv = bp[0];
  if (tid < 128) {
    float s = bpv;
#pragma unroll 8
    for (int n = 0; n < 128; ++n) s += ctx[tid * CLD + n] * wps[n];
    const float p = 1.0f / (1.0f + __expf(-s));
    pbuf[tid] = p;
    cpred[(size_t)(m0 + tid) * K_DIM + kb] = p;
  }
  __syncthreads();

  // ---- blend: c_emb = pos*p + neg*(1-p), coalesced stores ----
#pragma unroll 4
  for (int it = 0; it < 32; ++it) {
    const int idx = tid + it * 256;   // 0..8191
    const int m = idx >> 6;
    const int e = idx & 63;
    const float p   = pbuf[m];
    const float pos = ctx[m * CLD + e];
    const float neg = ctx[m * CLD + 64 + e];
    cemb[((size_t)(m0 + m) * K_DIM + kb) * E_DIM + e] = pos * p + (1.0f - p) * neg;
  }
}

extern "C" void kernel_launch(void* const* d_in, const int* in_sizes, int n_in,
                              void* d_out, int out_size, void* d_ws, size_t ws_size,
                              hipStream_t stream) {
  const float* x  = (const float*)d_in[0];
  const float* W  = (const float*)d_in[1];
  const float* b  = (const float*)d_in[2];
  const float* wp = (const float*)d_in[3];
  const float* bp = (const float*)d_in[4];
  float* out   = (float*)d_out;
  float* cemb  = out;                                      // (B, K, E) flat
  float* cpred = out + (size_t)B_DIM * K_DIM * E_DIM;      // (B, K) flat

  dim3 grid(B_DIM / 128, K_DIM);
  dim3 block(256);
  hipLaunchKernelGGL(concept_embed_kernel, grid, block, 0, stream,
                     x, W, b, wp, bp, cemb, cpred);
}
